// LocalitySelfAttention_2482491097847
// MI455X (gfx1250) — matrix operand order repaired and checked by reading, not compiled, not run on hardware
//
#include <hip/hip_runtime.h>

#define BATCH  8
#define NTOK   2304
#define GRIDW  48
#define EMBED  512
#define NHEAD  8
#define HDIM   64
#define QKVC   1536
#define MTOT   (BATCH*NTOK)   // 18432

typedef __attribute__((ext_vector_type(16))) __bf16 v16bf;
typedef __attribute__((ext_vector_type(8)))  float  v8f;

struct alignas(16) U4 { unsigned int x, y, z, w; };
struct alignas(16) F4 { float x, y, z, w; };

union FragB16 { U4 u[2]; v16bf v; __bf16 e[16]; };

__device__ __forceinline__ unsigned short f2bf_u(float f) {
  __bf16 h = (__bf16)f;
  return __builtin_bit_cast(unsigned short, h);
}

// ---------------------------------------------------------------------------
// fp32 -> bf16 elementwise convert (used for x and both weight matrices;
// torch (out,in) weight layout is already the WMMA B-fragment layout).
// ---------------------------------------------------------------------------
__global__ void cvt_f32_bf16(const float* __restrict__ src,
                             unsigned short* __restrict__ dst, int n) {
  int i = blockIdx.x * blockDim.x + threadIdx.x;
  if (i < n) dst[i] = f2bf_u(src[i]);
}

// ---------------------------------------------------------------------------
// GEMM: out(M x ncols) = A(M x 512 bf16 row-major) * W(ncols x 512 bf16)^T
// Block tile 64 x 128, 8 waves of 32x32, K-chunks of 32, WMMA bf16 -> f32.
// A staged in LDS (4x reuse); W fragments loaded directly from global
// (weights are L2-resident; each lane's chunk is 16 contiguous bf16).
// mode 0: scatter into q[bhnd], k[bhnd], vT[bhdn] (bf16)
// mode 1: out[m*512+g] = acc + bias[g]  (fp32, final output)
// ---------------------------------------------------------------------------
__global__ __launch_bounds__(256)
void gemm_bf16(const unsigned short* __restrict__ Au,
               const unsigned short* __restrict__ Wu,
               int ncols, int mode,
               unsigned short* __restrict__ qp,
               unsigned short* __restrict__ kp,
               unsigned short* __restrict__ vTp,
               float* __restrict__ outp,
               const float* __restrict__ bias) {
  __shared__ __bf16 ldsA[64 * 40];     // 64 rows x 32 K, stride 40 (pad)

  const __bf16* A = reinterpret_cast<const __bf16*>(Au);
  const __bf16* W = reinterpret_cast<const __bf16*>(Wu);

  const int nBase = blockIdx.x * 128;
  const int mBase = blockIdx.y * 64;
  const int t = threadIdx.x;
  const int wave = t >> 5, lane = t & 31, lo = lane & 15, hi = lane >> 4;
  const int wm = wave >> 2, wn = wave & 3;

  v8f acc[2][2];
  for (int i = 0; i < 2; ++i)
    for (int j = 0; j < 2; ++j)
      for (int e = 0; e < 8; ++e) acc[i][j][e] = 0.0f;

  const int arow = t >> 2, acol = (t & 3) * 8;       // A-tile cooperative load
  // W fragment bases: row (output col) = nBase + wn*32 + j*16 + lo
  const __bf16* wrow0 = W + (size_t)(nBase + wn * 32 + lo) * EMBED;
  const __bf16* wrow1 = W + (size_t)(nBase + wn * 32 + 16 + lo) * EMBED;

  for (int kc = 0; kc < 16; ++kc) {
    const int k0 = kc * 32;
    // stage A tile (64x32) into LDS
    {
      const __bf16* gp = A + (size_t)(mBase + arow) * EMBED + k0 + acol;
      U4 val = *reinterpret_cast<const U4*>(gp);
      *reinterpret_cast<U4*>(&ldsA[arow * 40 + acol]) = val;
      if (kc + 1 < 16) __builtin_prefetch(gp + 32, 0, 1);
    }
    __syncthreads();

    FragB16 a[2], b[2];
    for (int i = 0; i < 2; ++i) {
      const int row = wm * 32 + i * 16 + lo;
      a[i].u[0] = *reinterpret_cast<const U4*>(&ldsA[row * 40 + hi * 16]);
      a[i].u[1] = *reinterpret_cast<const U4*>(&ldsA[row * 40 + hi * 16 + 8]);
    }
    // b[j].e[i] = W[nBase+wn*32+j*16+lo][k0 + hi*16 + i]  (contiguous 32 B)
    b[0].u[0] = *reinterpret_cast<const U4*>(wrow0 + k0 + hi * 16);
    b[0].u[1] = *reinterpret_cast<const U4*>(wrow0 + k0 + hi * 16 + 8);
    b[1].u[0] = *reinterpret_cast<const U4*>(wrow1 + k0 + hi * 16);
    b[1].u[1] = *reinterpret_cast<const U4*>(wrow1 + k0 + hi * 16 + 8);

    for (int i = 0; i < 2; ++i)
      for (int j = 0; j < 2; ++j)
        acc[i][j] = __builtin_amdgcn_wmma_f32_16x16x32_bf16(
            false, a[i].v, false, b[j].v, (short)0, acc[i][j], false, false);
    __syncthreads();
  }

  // epilogue (batch boundaries align with 64-row blocks: 2304 % 64 == 0)
  const int bi = mBase / NTOK;
  const int nb = mBase - bi * NTOK;
  for (int i = 0; i < 2; ++i)
    for (int j = 0; j < 2; ++j) {
      const int g = nBase + wn * 32 + j * 16 + lo;
      for (int vr = 0; vr < 8; ++vr) {
        const int mloc = wm * 32 + i * 16 + hi * 8 + vr;
        const float val = acc[i][j][vr];
        if (mode == 0) {
          const int s = g >> 9, rem = g & 511, h = rem >> 6, d = rem & 63;
          const int n = nb + mloc;
          if (s == 0)
            qp[(((size_t)bi * NHEAD + h) * NTOK + n) * HDIM + d] = f2bf_u(val);
          else if (s == 1)
            kp[(((size_t)bi * NHEAD + h) * NTOK + n) * HDIM + d] = f2bf_u(val);
          else
            vTp[(((size_t)bi * NHEAD + h) * HDIM + d) * NTOK + n] = f2bf_u(val);
        } else {
          outp[(size_t)(mBase + mloc) * EMBED + g] = val + bias[g];
        }
      }
    }
}

// ---------------------------------------------------------------------------
// Attention: per (b,h, 16-query tile). Full score row kept in LDS (fp32),
// single-pass softmax (no online rescale), then P@V split across waves.
// ---------------------------------------------------------------------------
__global__ __launch_bounds__(256)
void attn_kernel(const unsigned short* __restrict__ qU,
                 const unsigned short* __restrict__ kU,
                 const unsigned short* __restrict__ vTU,
                 unsigned short* __restrict__ aoU,
                 const float* __restrict__ temp,
                 const float* __restrict__ locw) {
  extern __shared__ float smem[];            // S[16][NTOK] + red[8][16][64]
  float* S   = smem;
  float* red = smem + 16 * NTOK;

  const __bf16* qm = reinterpret_cast<const __bf16*>(qU);
  const __bf16* km = reinterpret_cast<const __bf16*>(kU);
  const __bf16* vm = reinterpret_cast<const __bf16*>(vTU);

  const int by = blockIdx.y;
  const int bi = by / NHEAD, h = by - bi * NHEAD;
  const int q0 = blockIdx.x * 16;
  const int t = threadIdx.x, wave = t >> 5, lane = t & 31, lo = lane & 15, hi = lane >> 4;

  const __bf16* qh  = qm + ((size_t)(bi * NHEAD + h)) * NTOK * HDIM;
  const __bf16* kh  = km + ((size_t)(bi * NHEAD + h)) * NTOK * HDIM;
  const __bf16* vTh = vm + ((size_t)(bi * NHEAD + h)) * HDIM * NTOK;

  const float scale = __expf(temp[h]);
  const float lw = locw[h];

  // Q fragments pinned in registers: rows q0..q0+15, D split 0..31 / 32..63
  FragB16 aq[2];
  for (int f = 0; f < 2; ++f) {
    const __bf16* base = qh + (size_t)(q0 + lo) * HDIM + f * 32;
    aq[f].u[0] = *reinterpret_cast<const U4*>(base + hi * 16);
    aq[f].u[1] = *reinterpret_cast<const U4*>(base + hi * 16 + 8);
  }

  // ---- phase 1: S = scale * Q K^T + lw * locality ----
  for (int kt = 0; kt < NTOK; kt += 128) {
    const int kb0 = kt + wave * 16;     // this wave's 16 keys
    FragB16 bk[2];
    for (int f = 0; f < 2; ++f) {
      const __bf16* base = kh + (size_t)(kb0 + lo) * HDIM + f * 32 + hi * 16;
      bk[f].u[0] = *reinterpret_cast<const U4*>(base);
      bk[f].u[1] = *reinterpret_cast<const U4*>(base + 8);
    }
    v8f accS;
    for (int e = 0; e < 8; ++e) accS[e] = 0.0f;
    accS = __builtin_amdgcn_wmma_f32_16x16x32_bf16(
        false, aq[0].v, false, bk[0].v, (short)0, accS, false, false);
    accS = __builtin_amdgcn_wmma_f32_16x16x32_bf16(
        false, aq[1].v, false, bk[1].v, (short)0, accS, false, false);

    const int kidx = kb0 + lo;
    const int ky = kidx / GRIDW, kx = kidx - ky * GRIDW;
    for (int vr = 0; vr < 8; ++vr) {
      const int qr = hi * 8 + vr;
      const int qidx = q0 + qr;
      const int qy = qidx / GRIDW, qx = qidx - qy * GRIDW;
      const float dy = (float)(qy - ky), dx = (float)(qx - kx);
      // linspace(0,1,48): step 1/47; max dist2 = 2 -> loc = -(dy^2+dx^2)/(2*47^2)
      const float loc = -(dy * dy + dx * dx) * (1.0f / (2.0f * 47.0f * 47.0f));
      S[qr * NTOK + kidx] = accS[vr] * scale + lw * loc;
    }
  }
  __syncthreads();

  // ---- phase 2: softmax per row (wave-parallel, wave32 shuffles) ----
  for (int rr = 0; rr < 2; ++rr) {
    const int r = wave + rr * 8;
    float* row = S + r * NTOK;
    float m = -1e30f;
    for (int i = lane; i < NTOK; i += 32) m = fmaxf(m, row[i]);
    for (int off = 16; off; off >>= 1) m = fmaxf(m, __shfl_xor(m, off, 32));
    float sum = 0.0f;
    for (int i = lane; i < NTOK; i += 32) {
      const float e = __expf(row[i] - m);
      row[i] = e;
      sum += e;
    }
    for (int off = 16; off; off >>= 1) sum += __shfl_xor(sum, off, 32);
    const float inv = 1.0f / sum;
    for (int i = lane; i < NTOK; i += 32) row[i] *= inv;
  }
  __syncthreads();

  // ---- phase 3: O = P @ V, K-dim split across 8 waves (288 keys each) ----
  v8f co[4];
  for (int j = 0; j < 4; ++j)
    for (int e = 0; e < 8; ++e) co[j][e] = 0.0f;
  const int kstart = wave * (NTOK / 8);
  for (int c = 0; c < NTOK / 8; c += 32) {
    const int kb0 = kstart + c;
    // P A-fragment: rows = 16 queries, K = 32 keys (fp32 LDS -> bf16)
    const float* prow = S + lo * NTOK + kb0;
    F4 c0 = *reinterpret_cast<const F4*>(prow + hi * 16);
    F4 c1 = *reinterpret_cast<const F4*>(prow + hi * 16 + 4);
    F4 c2 = *reinterpret_cast<const F4*>(prow + hi * 16 + 8);
    F4 c3 = *reinterpret_cast<const F4*>(prow + hi * 16 + 12);
    float pe[16] = {c0.x, c0.y, c0.z, c0.w, c1.x, c1.y, c1.z, c1.w,
                    c2.x, c2.y, c2.z, c2.w, c3.x, c3.y, c3.z, c3.w};
    v16bf pa;
    for (int i = 0; i < 16; ++i) pa[i] = (__bf16)pe[i];

    for (int j = 0; j < 4; ++j) {
      FragB16 bv;  // V B-fragment: vT is [d][n] -> contiguous keys per lane
      const __bf16* base = vTh + (size_t)(j * 16 + lo) * NTOK + kb0 + hi * 16;
      bv.u[0] = *reinterpret_cast<const U4*>(base);
      bv.u[1] = *reinterpret_cast<const U4*>(base + 8);
      co[j] = __builtin_amdgcn_wmma_f32_16x16x32_bf16(
          false, pa, false, bv.v, (short)0, co[j], false, false);
    }
  }
  // partials -> LDS
  for (int j = 0; j < 4; ++j)
    for (int vr = 0; vr < 8; ++vr) {
      const int qr = hi * 8 + vr, d = j * 16 + lo;
      red[(wave * 16 + qr) * 64 + d] = co[j][vr];
    }
  __syncthreads();

  // tree-reduce 8 partials, write attn-out (bf16, [b][n][h*64+d] layout)
  __bf16* ao = reinterpret_cast<__bf16*>(aoU);
  for (int e = t; e < 16 * 64; e += 256) {
    const int qr = e >> 6, d = e & 63;
    float s = 0.0f;
    for (int w = 0; w < 8; ++w) s += red[(w * 16 + qr) * 64 + d];
    ao[((size_t)(bi * NTOK + q0 + qr)) * EMBED + h * HDIM + d] = (__bf16)s;
  }
}

// ---------------------------------------------------------------------------
extern "C" void kernel_launch(void* const* d_in, const int* in_sizes, int n_in,
                              void* d_out, int out_size, void* d_ws, size_t ws_size,
                              hipStream_t stream) {
  const float* x      = (const float*)d_in[0];
  const float* qkv_w  = (const float*)d_in[1];
  const float* proj_w = (const float*)d_in[2];
  const float* proj_b = (const float*)d_in[3];
  const float* temp   = (const float*)d_in[4];
  const float* locw   = (const float*)d_in[5];
  float* out = (float*)d_out;

  char* ws = (char*)d_ws;
  unsigned short* x_bf   = (unsigned short*)(ws + 0);          // 18,874,368 B
  unsigned short* wqkv   = (unsigned short*)(ws + 18874368);   //  1,572,864 B
  unsigned short* wproj  = (unsigned short*)(ws + 20447232);   //    524,288 B
  unsigned short* qb     = (unsigned short*)(ws + 20971520);   // 18,874,368 B
  unsigned short* kb     = (unsigned short*)(ws + 39845888);   // 18,874,368 B
  unsigned short* vTb    = (unsigned short*)(ws + 58720256);   // 18,874,368 B
  unsigned short* aob    = (unsigned short*)(ws + 77594624);   // 18,874,368 B

  // 1) bf16 conversions (weights kept in native (out,in) layout — that IS the
  //    WMMA B-fragment layout, no transpose needed)
  cvt_f32_bf16<<<(MTOT * EMBED) / 256, 256, 0, stream>>>(x, x_bf, MTOT * EMBED);
  cvt_f32_bf16<<<(QKVC * EMBED) / 256, 256, 0, stream>>>(qkv_w, wqkv, QKVC * EMBED);
  cvt_f32_bf16<<<(EMBED * EMBED) / 256, 256, 0, stream>>>(proj_w, wproj, EMBED * EMBED);

  // 2) QKV projection, scatter to q/k/vT head layouts
  gemm_bf16<<<dim3(QKVC / 128, MTOT / 64), 256, 0, stream>>>(
      x_bf, wqkv, QKVC, 0, qb, kb, vTb, nullptr, nullptr);

  // 3) locality-biased attention (147 KB score slab + 32 KB reduce in LDS)
  const size_t attn_lds = (size_t)16 * NTOK * 4 + (size_t)8 * 16 * 64 * 4; // 180224
  attn_kernel<<<dim3(NTOK / 16, BATCH * NHEAD), 256, attn_lds, stream>>>(
      qb, kb, vTb, aob, temp, locw);

  // 4) output projection + bias -> fp32 result
  gemm_bf16<<<dim3(EMBED / 128, MTOT / 64), 256, 0, stream>>>(
      aob, wproj, EMBED, 1, nullptr, nullptr, nullptr, out, proj_b);
}
